// Tacotron_37056977829938
// MI455X (gfx1250) — matrix operand order repaired and checked by reading, not compiled, not run on hardware
//
#include <hip/hip_runtime.h>
#include <hip/hip_bf16.h>

typedef __attribute__((ext_vector_type(16))) _Float16 v16h;
typedef __attribute__((ext_vector_type(8)))  _Float16 v8h;
typedef __attribute__((ext_vector_type(8)))  float    v8f;

// ---------------------------------------------------------------- constants
#define NB      16            // batch
#define TTEXT_  192
#define STEPS_  800
#define RVAL_   2
#define NSTEPS_ 400           // STEPS_/RVAL_
#define NMELS_  80
#define DEC_    256
#define LSTM_   1024
#define EMBED_  256
#define FFT_    1025
#define MAXR_   20

#define MEL_N   (NB*NMELS_*STEPS_)            // 1,024,000
#define LIN_N   (NB*FFT_*STEPS_)              // 13,120,000
#define ATTN_OFF (MEL_N + LIN_N)              // 14,144,000

enum { ACT_NONE = 0, ACT_RELU = 1, ACT_SIG = 2 };

__device__ __forceinline__ float sigf(float x) { return 1.f / (1.f + __expf(-x)); }
__device__ __forceinline__ int   imin(int a, int b) { return a < b ? a : b; }

__device__ __forceinline__ v16h pack16(v8h lo, v8h hi)
{
    return __builtin_shufflevector(lo, hi, 0, 1, 2, 3, 4, 5, 6, 7,
                                   8, 9, 10, 11, 12, 13, 14, 15);
}

// ---------------------------------------------------------------- WMMA K-block
// One 16x16x32 WMMA accumulate.
//  arow_ptr : A row base for this lane (A + arow*lda), aok = row valid
//  wrow_ptr : W row base for this lane (W + wrow*K),  wok = row valid
// A layout (ISA 7.12.2, 16-bit A 16x32):
//   lanes 0-15 : elems 0..7 = K kbase+0..7 , elems 8..15 = K kbase+16..23
//   lanes 16-31: elems 0..7 = K kbase+8..15, elems 8..15 = K kbase+24..31
// B layout (16-bit B 32x16): same K order as A (lanes 0-15 K kbase+0..7,16..23; lanes 16-31 K kbase+8..15,24..31);
//   B[k][col] == W[col][k] -> two 8-half chunks of one W row, 16 halves apart.
// All K values in this model are multiples of 8, and every 8-half chunk starts at a
// multiple of 8 -> chunks are all-valid or all-invalid; TAIL=true adds chunk guards.
template <bool TAIL>
__device__ __forceinline__ void wmma_kblock(const _Float16* __restrict__ arow_ptr, bool aok,
                                            const _Float16* __restrict__ wrow_ptr, bool wok,
                                            int kbase, int K, v8f& acc)
{
    const int half = (threadIdx.x >> 4) & 1;
    const int ao0 = kbase + half * 8;
    const int ao1 = kbase + 16 + half * 8;
    const int bo  = kbase + half * 8;

    v16h a = {};
    v16h b = {};
    if (aok) {
        v8h lo = {}, hi = {};
        if (!TAIL || (ao0 + 8 <= K)) lo = *(const v8h*)(arow_ptr + ao0);
        if (!TAIL || (ao1 + 8 <= K)) hi = *(const v8h*)(arow_ptr + ao1);
        a = pack16(lo, hi);
    }
    if (wok) {
        v8h lo = {}, hi = {};
        if (!TAIL || (bo + 8  <= K)) lo = *(const v8h*)(wrow_ptr + bo);
        if (!TAIL || (bo + 24 <= K)) hi = *(const v8h*)(wrow_ptr + bo + 16);
        b = pack16(lo, hi);
    }
    acc = __builtin_amdgcn_wmma_f32_16x16x32_f16(false, a, false, b, (short)0, acc,
                                                 false, false);
}

// ---------------------------------------------------------------- generic GEMM
// C[M,N] = act(A[M,K] @ W[N,K]^T + bias)     (A,W fp16; C fp32; optional fp16 copy)
// EDGE=false: M%16==0 && N%16==0 -> no lane guards anywhere (pure b128+wmma stream).
// EDGE=true : ragged M/N tiles keep per-lane row/col guards (post_proj N=1025 only).
template <bool EDGE>
__global__ void k_gemm(const _Float16* __restrict__ A, int lda,
                       const _Float16* __restrict__ W,
                       const float* __restrict__ bias,
                       float* __restrict__ C, int ldc,
                       _Float16* __restrict__ C16, int ldc16,
                       int M, int N, int K, int act)
{
    __builtin_amdgcn_s_wait_tensorcnt(0);
    const int wave  = threadIdx.x >> 5;
    const int lane  = threadIdx.x & 31;
    const int l16   = lane & 15;
    const int mtile = blockIdx.x;
    const int ntile = blockIdx.y * 8 + wave;
    const int arow0 = mtile * 16;
    const int ncol0 = ntile * 16;
    if (ncol0 >= N) return;

    const int  arow = arow0 + l16;
    const int  wrow = ncol0 + l16;
    const bool aok  = EDGE ? (arow < M) : true;
    const bool wok  = EDGE ? (wrow < N) : true;
    const _Float16* arow_ptr = A + (size_t)(EDGE ? imin(arow, M - 1) : arow) * lda;
    const _Float16* wrow_ptr = W + (size_t)(EDGE ? imin(wrow, N - 1) : wrow) * K;

    v8f acc = {};
    const int K32 = K & ~31;
    for (int kb = 0; kb < K32; kb += 32) {
        __builtin_prefetch((const void*)(wrow_ptr + kb + 64), 0, 3);
        wmma_kblock<false>(arow_ptr, aok, wrow_ptr, wok, kb, K, acc);
    }
    if (K32 < K)
        wmma_kblock<true>(arow_ptr, aok, wrow_ptr, wok, K32, K, acc);

    const int col = ncol0 + l16;
    const int rbase = arow0 + ((lane >> 4) << 3);
    if (!EDGE) {
        const float bv = (bias != nullptr) ? bias[col] : 0.f;
#pragma unroll
        for (int r = 0; r < 8; ++r) {
            float v = acc[r] + bv;
            if (act == ACT_RELU) v = fmaxf(v, 0.f);
            else if (act == ACT_SIG) v = sigf(v);
            C[(size_t)(rbase + r) * ldc + col] = v;
            if (C16) C16[(size_t)(rbase + r) * ldc16 + col] = (_Float16)v;
        }
    } else {
        const float bv = (bias != nullptr && col < N) ? bias[col] : 0.f;
        if (col < N) {
#pragma unroll
            for (int r = 0; r < 8; ++r) {
                if (rbase + r < M) {
                    float v = acc[r] + bv;
                    if (act == ACT_RELU) v = fmaxf(v, 0.f);
                    else if (act == ACT_SIG) v = sigf(v);
                    C[(size_t)(rbase + r) * ldc + col] = v;
                    if (C16) C16[(size_t)(rbase + r) * ldc16 + col] = (_Float16)v;
                }
            }
        }
    }
}

// ---------------------------------------------------------------- fused biGRU (ch=128)
// One workgroup per direction (grid=2). h kept in LDS as f16; per-step GEMMs
// 16x384x128 via WMMA (A fragments ds_load_b128, weights global b128, L2-resident).
#define GCH 128
__global__ void k_gru_seq(const _Float16* __restrict__ xseq,   // [T,16,GCH] f16
                          const _Float16* __restrict__ Wih_f, const _Float16* __restrict__ Whh_f,
                          const float* __restrict__ bih_f, const float* __restrict__ bhh_f,
                          const _Float16* __restrict__ Wih_b, const _Float16* __restrict__ Whh_b,
                          const float* __restrict__ bih_b, const float* __restrict__ bhh_b,
                          float* __restrict__ out, _Float16* __restrict__ out16,
                          int T, int strideT, size_t strideB)
{
    __shared__ float    sgi[16 * 3 * GCH];
    __shared__ float    sgh[16 * 3 * GCH];
    __shared__ _Float16 sh16[16 * GCH];

    const int reverse = blockIdx.x;                    // 0 = fwd, 1 = bwd
    const int outOff  = reverse ? GCH : 0;
    const _Float16* Wih = reverse ? Wih_b : Wih_f;
    const _Float16* Whh = reverse ? Whh_b : Whh_f;
    const float*    bih = reverse ? bih_b : bih_f;
    const float*    bhh = reverse ? bhh_b : bhh_f;

    const int tid  = threadIdx.x;
    const int lane = tid & 31;
    const int l16  = lane & 15;
    const int wave = tid >> 5;

    for (int i = tid; i < 16 * GCH; i += blockDim.x) sh16[i] = (_Float16)0.f;
    __syncthreads();

    const _Float16* hrow = sh16 + (size_t)l16 * GCH;   // LDS row for this lane

    const int NT = 3 * GCH / 16;        // 24 N-tiles
    for (int s = 0; s < T; ++s) {
        const int t = reverse ? (T - 1 - s) : s;
        const _Float16* xrow = xseq + ((size_t)t * 16 + l16) * GCH;

        for (int nt = wave; nt < NT; nt += 8) {
            const _Float16* wih_row = Wih + (size_t)(nt * 16 + l16) * GCH;
            const _Float16* whh_row = Whh + (size_t)(nt * 16 + l16) * GCH;
            v8f gi = {}, gh = {};
#pragma unroll
            for (int kb = 0; kb < GCH; kb += 32) {
                wmma_kblock<false>(xrow, true, wih_row, true, kb, GCH, gi);
                wmma_kblock<false>(hrow, true, whh_row, true, kb, GCH, gh);
            }
            const int col = nt * 16 + l16;
#pragma unroll
            for (int r = 0; r < 8; ++r) {
                int row = r + ((lane >> 4) << 3);
                sgi[row * 3 * GCH + col] = gi[r] + bih[col];
                sgh[row * 3 * GCH + col] = gh[r] + bhh[col];
            }
        }
        __syncthreads();

        for (int i = tid; i < 16 * GCH; i += blockDim.x) {
            const int b = i / GCH, c = i % GCH;
            const float* gib = sgi + b * 3 * GCH;
            const float* ghb = sgh + b * 3 * GCH;
            float r_ = sigf(gib[c] + ghb[c]);
            float z_ = sigf(gib[GCH + c] + ghb[GCH + c]);
            float n_ = tanhf(gib[2 * GCH + c] + r_ * ghb[2 * GCH + c]);
            float h_ = (1.f - z_) * n_ + z_ * (float)sh16[i];
            size_t o = (size_t)t * strideT + (size_t)b * strideB + outOff + c;
            out[o] = h_;
            out16[o] = (_Float16)h_;
            sh16[i] = (_Float16)h_;
        }
        __syncthreads();
    }
}

// ---------------------------------------------------------------- small kernels
__global__ void k_f32_to_f16(const float* __restrict__ s, _Float16* __restrict__ d, int n)
{
    int i = blockIdx.x * blockDim.x + threadIdx.x;
    if (i < n) d[i] = (_Float16)s[i];
}

__global__ void k_zero16(_Float16* p, int n)
{
    int i = blockIdx.x * blockDim.x + threadIdx.x;
    if (i < n) p[i] = (_Float16)0.f;
}
__global__ void k_zero32(float* p, int n)
{
    int i = blockIdx.x * blockDim.x + threadIdx.x;
    if (i < n) p[i] = 0.f;
}

__global__ void k_embed(const int* __restrict__ x, const float* __restrict__ emb,
                        _Float16* __restrict__ out, int n, int E)
{
    int i = blockIdx.x * blockDim.x + threadIdx.x;
    if (i >= n) return;
    int e = i % E, bt = i / E;
    out[i] = (_Float16)emb[(size_t)x[bt] * E + e];
}

__global__ void k_conv1d(const float* __restrict__ x, const float* __restrict__ w,
                         float* __restrict__ y,
                         int Cin, int L, int Cout, int ks, int pad, int relu,
                         int ocOff, int ocStride)
{
    int i = blockIdx.x * blockDim.x + threadIdx.x;
    if (i >= NB * Cout * L) return;
    int t  = i % L;
    int co = (i / L) % Cout;
    int b  = i / (L * Cout);
    float acc = 0.f;
    for (int ci = 0; ci < Cin; ++ci) {
        const float* xr = x + ((size_t)b * Cin + ci) * L;
        const float* wr = w + ((size_t)co * Cin + ci) * ks;
        for (int kk = 0; kk < ks; ++kk) {
            int xi = t - pad + kk;
            if (xi >= 0 && xi < L) acc += xr[xi] * wr[kk];
        }
    }
    if (relu) acc = fmaxf(acc, 0.f);
    y[((size_t)b * ocStride + ocOff + co) * L + t] = acc;
}

__global__ void k_bn_stats(const float* __restrict__ x, int L, int cStride, int cOff,
                           float* __restrict__ mv)
{
    int c = blockIdx.x;
    __shared__ float s1[256], s2[256];
    float a = 0.f, q = 0.f;
    for (int i = threadIdx.x; i < NB * L; i += 256) {
        int b = i / L, t = i % L;
        float v = x[((size_t)b * cStride + cOff + c) * L + t];
        a += v; q += v * v;
    }
    s1[threadIdx.x] = a; s2[threadIdx.x] = q;
    __syncthreads();
    for (int s = 128; s > 0; s >>= 1) {
        if (threadIdx.x < s) { s1[threadIdx.x] += s1[threadIdx.x + s]; s2[threadIdx.x] += s2[threadIdx.x + s]; }
        __syncthreads();
    }
    if (threadIdx.x == 0) {
        float n = (float)(NB * L);
        float mean = s1[0] / n;
        float var  = s2[0] / n - mean * mean;
        mv[c * 2] = mean;
        mv[c * 2 + 1] = rsqrtf(var + 1e-5f);
    }
}

__global__ void k_bn_apply(float* __restrict__ x, int L, int cStride, int cOff, int C,
                           const float* __restrict__ g, const float* __restrict__ beta,
                           const float* __restrict__ mv)
{
    int i = blockIdx.x * blockDim.x + threadIdx.x;
    if (i >= NB * C * L) return;
    int t = i % L;
    int c = (i / L) % C;
    int b = i / (L * C);
    size_t off = ((size_t)b * cStride + cOff + c) * L + t;
    x[off] = (x[off] - mv[c * 2]) * mv[c * 2 + 1] * g[c] + beta[c];
}

__global__ void k_maxpool(const float* __restrict__ x, float* __restrict__ y, int C, int L)
{
    int i = blockIdx.x * blockDim.x + threadIdx.x;
    if (i >= NB * C * L) return;
    int t = i % L;
    size_t bc = i / L;
    float v = x[bc * L + t];
    if (t > 0) v = fmaxf(v, x[bc * L + t - 1]);
    y[bc * L + t] = v;
}

__global__ void k_bcl_to_btc(const float* __restrict__ x, float* __restrict__ y32,
                             _Float16* __restrict__ y16, int C, int L)
{
    int i = blockIdx.x * blockDim.x + threadIdx.x;
    if (i >= NB * C * L) return;
    int c = i % C;
    int t = (i / C) % L;
    int b = i / (C * L);
    float v = x[((size_t)b * C + c) * L + t];
    size_t o = ((size_t)b * L + t) * C + c;
    y32[o] = v;
    y16[o] = (_Float16)v;
}

__global__ void k_btc_to_bcl(const float* __restrict__ x, float* __restrict__ y, int C, int L)
{
    int i = blockIdx.x * blockDim.x + threadIdx.x;
    if (i >= NB * C * L) return;
    int c = i % C;
    int t = (i / C) % L;
    int b = i / (C * L);
    y[((size_t)b * C + c) * L + t] = x[((size_t)b * L + t) * C + c];
}

__global__ void k_btc_to_tbc16(const _Float16* __restrict__ x, _Float16* __restrict__ y,
                               int C, int L)
{
    int i = blockIdx.x * blockDim.x + threadIdx.x;
    if (i >= NB * C * L) return;
    int c = i % C;
    int t = (i / C) % L;
    int b = i / (C * L);
    y[((size_t)t * NB + b) * C + c] = x[((size_t)b * L + t) * C + c];
}

__global__ void k_highway(const float* __restrict__ h1, const float* __restrict__ hg,
                          float* __restrict__ x32, _Float16* __restrict__ x16, int n)
{
    int i = blockIdx.x * blockDim.x + threadIdx.x;
    if (i >= n) return;
    float g = hg[i];
    float v = g * h1[i] + (1.f - g) * x32[i];
    x32[i] = v;
    x16[i] = (_Float16)v;
}

__global__ void k_addin(float* __restrict__ a, const float* __restrict__ b, int n)
{
    int i = blockIdx.x * blockDim.x + threadIdx.x;
    if (i < n) a[i] += b[i];
}

__global__ void k_add(const float* __restrict__ a, const float* __restrict__ b,
                      float* __restrict__ o32, _Float16* __restrict__ o16, int n)
{
    int i = blockIdx.x * blockDim.x + threadIdx.x;
    if (i >= n) return;
    float v = a[i] + b[i];
    o32[i] = v;
    o16[i] = (_Float16)v;
}

__global__ void k_gru_cell(const float* __restrict__ gi, const float* __restrict__ gh,
                           float* __restrict__ h32,
                           _Float16* __restrict__ h16a, int lda,
                           _Float16* __restrict__ h16b, int ldb, int H)
{
    int i = blockIdx.x * blockDim.x + threadIdx.x;
    if (i >= NB * H) return;
    int b = i / H, c = i % H;
    const float* gib = gi + (size_t)b * 3 * H;
    const float* ghb = gh + (size_t)b * 3 * H;
    float r_ = sigf(gib[c] + ghb[c]);
    float z_ = sigf(gib[H + c] + ghb[H + c]);
    float n_ = tanhf(gib[2 * H + c] + r_ * ghb[2 * H + c]);
    float hv = (1.f - z_) * n_ + z_ * h32[i];
    h32[i] = hv;
    h16a[(size_t)b * lda + c] = (_Float16)hv;
    if (h16b) h16b[(size_t)b * ldb + c] = (_Float16)hv;
}

__global__ void k_lstm(const float* __restrict__ gx, const float* __restrict__ gh,
                       float* __restrict__ c32, float* __restrict__ h32,
                       _Float16* __restrict__ h16, int H)
{
    int i = blockIdx.x * blockDim.x + threadIdx.x;
    if (i >= NB * H) return;
    int b = i / H, c = i % H;
    const float* a = gx + (size_t)b * 4 * H;
    const float* d = gh + (size_t)b * 4 * H;
    float ii = sigf(a[c] + d[c]);
    float ff = sigf(a[H + c] + d[H + c]);
    float gg = tanhf(a[2 * H + c] + d[2 * H + c]);
    float oo = sigf(a[3 * H + c] + d[3 * H + c]);
    float cv = ff * c32[i] + ii * gg;
    c32[i] = cv;
    float hv = oo * tanhf(cv);
    h32[i] = hv;
    h16[i] = (_Float16)hv;
}

__global__ void k_frame(const float* __restrict__ m, _Float16* __restrict__ pin,
                        int col, int zero)
{
    int i = blockIdx.x * blockDim.x + threadIdx.x;
    if (i >= NB * NMELS_) return;
    int b = i / NMELS_, n = i % NMELS_;
    float v = zero ? 0.f : m[((size_t)b * NMELS_ + n) * STEPS_ + col];
    pin[i] = (_Float16)v;
}

__global__ void k_attn_score(const float* __restrict__ ep, const float* __restrict__ q,
                             const float* __restrict__ v, float* __restrict__ u,
                             int T, int D)
{
    int t = blockIdx.x, b = blockIdx.y;
    __shared__ float red[256];
    int d = threadIdx.x;
    float val = 0.f;
    if (d < D) val = tanhf(ep[((size_t)b * T + t) * D + d] + q[(size_t)b * D + d]) * v[d];
    red[d] = val;
    __syncthreads();
    for (int s = 128; s > 0; s >>= 1) {
        if (d < s) red[d] += red[d + s];
        __syncthreads();
    }
    if (d == 0) u[b * T + t] = red[0];
}

__global__ void k_softmax(const float* __restrict__ u, float* __restrict__ sc,
                          float* __restrict__ attn_out, int T, size_t strideB)
{
    int b = blockIdx.x;
    int t = threadIdx.x;
    __shared__ float red[256];
    float v = (t < T) ? u[b * T + t] : -1e30f;
    red[t] = v;
    __syncthreads();
    for (int s = 128; s > 0; s >>= 1) {
        if (t < s) red[t] = fmaxf(red[t], red[t + s]);
        __syncthreads();
    }
    float mx = red[0];
    __syncthreads();
    float e = (t < T) ? __expf(v - mx) : 0.f;
    red[t] = e;
    __syncthreads();
    for (int s = 128; s > 0; s >>= 1) {
        if (t < s) red[t] += red[t + s];
        __syncthreads();
    }
    float inv = 1.f / red[0];
    if (t < T) {
        float s_ = e * inv;
        sc[b * T + t] = s_;
        attn_out[(size_t)b * strideB + t] = s_;
    }
}

__global__ void k_context(const float* __restrict__ sc, const float* __restrict__ enc,
                          _Float16* __restrict__ d1, int ld1,
                          _Float16* __restrict__ d2, int ld2, int T, int D)
{
    int b = blockIdx.x, d = threadIdx.x;
    if (d >= D) return;
    float acc = 0.f;
    for (int t = 0; t < T; ++t)
        acc += sc[b * T + t] * enc[((size_t)b * T + t) * D + d];
    d1[(size_t)b * ld1 + d] = (_Float16)acc;
    d2[(size_t)b * ld2 + d] = (_Float16)acc;
}

__global__ void k_write_mel(const float* __restrict__ mel, float* __restrict__ out, int step)
{
    int i = blockIdx.x * blockDim.x + threadIdx.x;
    if (i >= NB * NMELS_ * RVAL_) return;
    int j = i % RVAL_;
    int n = (i / RVAL_) % NMELS_;
    int b = i / (RVAL_ * NMELS_);
    out[(size_t)b * NMELS_ * STEPS_ + (size_t)n * STEPS_ + step * RVAL_ + j] =
        mel[(size_t)b * (NMELS_ * MAXR_) + n * MAXR_ + j];
}

__global__ void k_lin_tr(const float* __restrict__ lin, float* __restrict__ out)
{
    size_t i = (size_t)blockIdx.x * blockDim.x + threadIdx.x;
    if (i >= (size_t)NB * STEPS_ * FFT_) return;
    int f = (int)(i % FFT_);
    size_t bt = i / FFT_;
    int t = (int)(bt % STEPS_);
    int b = (int)(bt / STEPS_);
    out[(size_t)b * FFT_ * STEPS_ + (size_t)f * STEPS_ + t] = lin[i];
}

// ---------------------------------------------------------------- host helpers
static inline int ceilDiv(int a, int b) { return (a + b - 1) / b; }

struct Bump {
    char* p;
    void* take(size_t b) { void* r = p; p += (b + 255) & ~(size_t)255; return r; }
};

static inline void gemm(hipStream_t s, const _Float16* A, int lda, const _Float16* W,
                        const float* bias, float* C, int ldc, _Float16* C16, int ldc16,
                        int M, int N, int K, int act)
{
    dim3 g(ceilDiv(M, 16), ceilDiv(N, 128));
    if ((M % 16) == 0 && (N % 16) == 0)
        k_gemm<false><<<g, 256, 0, s>>>(A, lda, W, bias, C, ldc, C16, ldc16, M, N, K, act);
    else
        k_gemm<true><<<g, 256, 0, s>>>(A, lda, W, bias, C, ldc, C16, ldc16, M, N, K, act);
}

struct BN { const float* g; const float* b; };
struct CbhgP {
    int K, Cin, proj1C, proj2C;
    const float* bank_w[16]; BN bank_bn[16];
    const float* proj1_w; BN bn1;
    const float* proj2_w; BN bn2;
    const _Float16* hw_W1[4]; const _Float16* hw_W2[4];
    const float* hw_b1[4]; const float* hw_b2[4];
    const _Float16* pre_highway;        // null if absent
    const _Float16* gWih_f; const _Float16* gWhh_f; const float* gbih_f; const float* gbhh_f;
    const _Float16* gWih_b; const _Float16* gWhh_b; const float* gbih_b; const float* gbhh_b;
};

// ws passed BY VALUE -> temporaries are released on return
static void run_cbhg(hipStream_t st, Bump ws, const CbhgP& P,
                     const float* x_bcl, int L, float* out32, _Float16* out16)
{
    const int ch = 128;
    const int CB = P.K * ch;

    float* cb = (float*)ws.take((size_t)NB * CB * L * 4);
    float* mv = (float*)ws.take(2 * 256 * 4);
    for (int k = 0; k < P.K; ++k) {
        int ks = k + 1, pad = ks / 2;
        int tot = NB * ch * L;
        k_conv1d<<<ceilDiv(tot, 256), 256, 0, st>>>(x_bcl, P.bank_w[k], cb,
                                                    P.Cin, L, ch, ks, pad, 1, k * ch, CB);
        k_bn_stats<<<ch, 256, 0, st>>>(cb, L, CB, k * ch, mv);
        k_bn_apply<<<ceilDiv(tot, 256), 256, 0, st>>>(cb, L, CB, k * ch, ch,
                                                      P.bank_bn[k].g, P.bank_bn[k].b, mv);
    }
    float* mp = (float*)ws.take((size_t)NB * CB * L * 4);
    k_maxpool<<<ceilDiv(NB * CB * L, 256), 256, 0, st>>>(cb, mp, CB, L);

    float* pr1 = (float*)ws.take((size_t)NB * P.proj1C * L * 4);
    k_conv1d<<<ceilDiv(NB * P.proj1C * L, 256), 256, 0, st>>>(mp, P.proj1_w, pr1,
                                                              CB, L, P.proj1C, 3, 1, 1, 0, P.proj1C);
    k_bn_stats<<<P.proj1C, 256, 0, st>>>(pr1, L, P.proj1C, 0, mv);
    k_bn_apply<<<ceilDiv(NB * P.proj1C * L, 256), 256, 0, st>>>(pr1, L, P.proj1C, 0, P.proj1C,
                                                                P.bn1.g, P.bn1.b, mv);
    float* pr2 = (float*)ws.take((size_t)NB * P.proj2C * L * 4);
    k_conv1d<<<ceilDiv(NB * P.proj2C * L, 256), 256, 0, st>>>(pr1, P.proj2_w, pr2,
                                                              P.proj1C, L, P.proj2C, 3, 1, 0, 0, P.proj2C);
    k_bn_stats<<<P.proj2C, 256, 0, st>>>(pr2, L, P.proj2C, 0, mv);
    k_bn_apply<<<ceilDiv(NB * P.proj2C * L, 256), 256, 0, st>>>(pr2, L, P.proj2C, 0, P.proj2C,
                                                                P.bn2.g, P.bn2.b, mv);
    k_addin<<<ceilDiv(NB * P.proj2C * L, 256), 256, 0, st>>>(pr2, x_bcl, NB * P.proj2C * L);

    const int BL = NB * L;
    float*    y32 = (float*)ws.take((size_t)BL * ch * 4);
    _Float16* y16 = (_Float16*)ws.take((size_t)BL * ch * 2);
    if (P.pre_highway) {
        float*    t32 = (float*)ws.take((size_t)BL * P.proj2C * 4);
        _Float16* t16 = (_Float16*)ws.take((size_t)BL * P.proj2C * 2);
        k_bcl_to_btc<<<ceilDiv(NB * P.proj2C * L, 256), 256, 0, st>>>(pr2, t32, t16, P.proj2C, L);
        gemm(st, t16, P.proj2C, P.pre_highway, nullptr, y32, ch, y16, ch, BL, ch, P.proj2C, ACT_NONE);
    } else {
        k_bcl_to_btc<<<ceilDiv(NB * ch * L, 256), 256, 0, st>>>(pr2, y32, y16, ch, L);
    }

    float* t1 = (float*)ws.take((size_t)BL * ch * 4);
    float* t2 = (float*)ws.take((size_t)BL * ch * 4);
    for (int h = 0; h < 4; ++h) {
        gemm(st, y16, ch, P.hw_W1[h], P.hw_b1[h], t1, ch, nullptr, 0, BL, ch, ch, ACT_RELU);
        gemm(st, y16, ch, P.hw_W2[h], P.hw_b2[h], t2, ch, nullptr, 0, BL, ch, ch, ACT_SIG);
        k_highway<<<ceilDiv(BL * ch, 256), 256, 0, st>>>(t1, t2, y32, y16, BL * ch);
    }

    _Float16* xseq = (_Float16*)ws.take((size_t)L * NB * ch * 2);
    k_btc_to_tbc16<<<ceilDiv(NB * ch * L, 256), 256, 0, st>>>(y16, xseq, ch, L);
    // out layout [B, L, 2ch]: off = b*L*2ch + t*2ch + c
    k_gru_seq<<<2, 256, 0, st>>>(xseq, P.gWih_f, P.gWhh_f, P.gbih_f, P.gbhh_f,
                                 P.gWih_b, P.gWhh_b, P.gbih_b, P.gbhh_b,
                                 out32, out16, L, 2 * ch, (size_t)L * 2 * ch);
}

// ---------------------------------------------------------------- launch
extern "C" void kernel_launch(void* const* d_in, const int* in_sizes, int n_in,
                              void* d_out, int out_size, void* d_ws, size_t ws_size,
                              hipStream_t stream)
{
    const int*   x = (const int*)d_in[0];
    const float* m = (const float*)d_in[1];
    // d_in[2] = r (known RVAL_=2)
    int ai = 3;
    auto F = [&]() { return (const float*)d_in[ai++]; };

    Bump ws{(char*)d_ws};
    auto cvt = [&](const float* s, size_t n) {
        _Float16* d = (_Float16*)ws.take(n * sizeof(_Float16));
        k_f32_to_f16<<<ceilDiv((int)n, 256), 256, 0, stream>>>(s, d, (int)n);
        return d;
    };

    // ---- params in jax-tree (alphabetical dict) flatten order ----
    // decoder
    _Float16* attnW16 = cvt(F(), 256 * 256);                       // attn_W
    _Float16* arWhh16 = cvt(F(), 768 * 256);                       // attn_rnn.Whh
    _Float16* arWih16 = cvt(F(), 768 * 384);                       // attn_rnn.Wih
    const float* ar_bhh = F();
    const float* ar_bih = F();
    const float* attn_v = F();                                     // attn_v [256]
    _Float16* l1Whh16 = cvt(F(), (size_t)4096 * 1024);             // lstm1
    _Float16* l1Wih16 = cvt(F(), (size_t)4096 * 1024);
    const float* l1_bhh = F();
    const float* l1_bih = F();
    _Float16* l2Whh16 = cvt(F(), (size_t)4096 * 1024);             // lstm2
    _Float16* l2Wih16 = cvt(F(), (size_t)4096 * 1024);
    const float* l2_bhh = F();
    const float* l2_bih = F();
    _Float16* melW16 = cvt(F(), (size_t)1600 * 1024);              // mel_W
    _Float16* dpW1 = cvt(F(), 256 * 80);                           // dec prenet W1
    _Float16* dpW2 = cvt(F(), 128 * 256);                          // dec prenet W2
    const float* dp_b1 = F();
    const float* dp_b2 = F();
    _Float16* rnninW16 = cvt(F(), (size_t)1024 * 512);             // rnn_in_W
    const float* rnnin_b = F();

    // encoder.cbhg
    CbhgP E{};
    E.K = 16; E.Cin = 128; E.proj1C = 128; E.proj2C = 128;
    for (int k = 0; k < 16; ++k) {
        E.bank_bn[k].b = F();
        E.bank_bn[k].g = F();
        E.bank_w[k]    = F();
    }
    E.gWhh_b = cvt(F(), 384 * 128);  E.gWih_b = cvt(F(), 384 * 128);   // gru.bwd
    E.gbhh_b = F();  E.gbih_b = F();
    E.gWhh_f = cvt(F(), 384 * 128);  E.gWih_f = cvt(F(), 384 * 128);   // gru.fwd
    E.gbhh_f = F();  E.gbih_f = F();
    for (int h = 0; h < 4; ++h) {
        E.hw_W1[h] = cvt(F(), 128 * 128);
        E.hw_W2[h] = cvt(F(), 128 * 128);
        E.hw_b1[h] = F();
        E.hw_b2[h] = F();
    }
    E.bn1.b = F(); E.bn1.g = F(); E.proj1_w = F();                 // proj1
    E.bn2.b = F(); E.bn2.g = F(); E.proj2_w = F();                 // proj2
    E.pre_highway = nullptr;

    const float* emb = F();                                        // encoder.embedding
    _Float16* epW1 = cvt(F(), 256 * 256);                          // encoder.prenet
    _Float16* epW2 = cvt(F(), 128 * 256);
    const float* ep_b1 = F();
    const float* ep_b2 = F();

    _Float16* encProj16  = cvt(F(), 256 * 256);                    // encoder_proj
    _Float16* postProj16 = cvt(F(), (size_t)1025 * 256);           // post_proj

    // postnet cbhg
    CbhgP Pn{};
    Pn.K = 8; Pn.Cin = 80; Pn.proj1C = 256; Pn.proj2C = 80;
    for (int k = 0; k < 8; ++k) {
        Pn.bank_bn[k].b = F();
        Pn.bank_bn[k].g = F();
        Pn.bank_w[k]    = F();
    }
    Pn.gWhh_b = cvt(F(), 384 * 128);  Pn.gWih_b = cvt(F(), 384 * 128);
    Pn.gbhh_b = F();  Pn.gbih_b = F();
    Pn.gWhh_f = cvt(F(), 384 * 128);  Pn.gWih_f = cvt(F(), 384 * 128);
    Pn.gbhh_f = F();  Pn.gbih_f = F();
    for (int h = 0; h < 4; ++h) {
        Pn.hw_W1[h] = cvt(F(), 128 * 128);
        Pn.hw_W2[h] = cvt(F(), 128 * 128);
        Pn.hw_b1[h] = F();
        Pn.hw_b2[h] = F();
    }
    Pn.pre_highway = cvt(F(), 128 * 80);
    Pn.bn1.b = F(); Pn.bn1.g = F(); Pn.proj1_w = F();
    Pn.bn2.b = F(); Pn.bn2.g = F(); Pn.proj2_w = F();

    // ---------------- encoder ----------------
    const int BT = NB * TTEXT_;
    _Float16* e16 = (_Float16*)ws.take((size_t)BT * EMBED_ * 2);
    k_embed<<<ceilDiv(BT * EMBED_, 256), 256, 0, stream>>>(x, emb, e16, BT * EMBED_, EMBED_);

    float*    p1_32 = (float*)ws.take((size_t)BT * 256 * 4);
    _Float16* p1_16 = (_Float16*)ws.take((size_t)BT * 256 * 2);
    gemm(stream, e16, 256, epW1, ep_b1, p1_32, 256, p1_16, 256, BT, 256, 256, ACT_RELU);
    float*    pre32 = (float*)ws.take((size_t)BT * 128 * 4);
    _Float16* pre16 = (_Float16*)ws.take((size_t)BT * 128 * 2);
    gemm(stream, p1_16, 256, epW2, ep_b2, pre32, 128, pre16, 128, BT, 128, 256, ACT_RELU);

    float* xbcl = (float*)ws.take((size_t)NB * 128 * TTEXT_ * 4);
    k_btc_to_bcl<<<ceilDiv(NB * 128 * TTEXT_, 256), 256, 0, stream>>>(pre32, xbcl, 128, TTEXT_);

    float*    enc32 = (float*)ws.take((size_t)BT * 256 * 4);
    _Float16* enc16 = (_Float16*)ws.take((size_t)BT * 256 * 2);
    run_cbhg(stream, ws, E, xbcl, TTEXT_, enc32, enc16);

    float* ep32 = (float*)ws.take((size_t)BT * 256 * 4);
    gemm(stream, enc16, 256, encProj16, nullptr, ep32, 256, nullptr, 0, BT, 256, 256, ACT_NONE);

    // ---------------- decoder buffers ----------------
    _Float16* pin16   = (_Float16*)ws.take(NB * NMELS_ * 2);
    float*    pp1_32  = (float*)ws.take(NB * 256 * 4);
    _Float16* pp1_16  = (_Float16*)ws.take(NB * 256 * 2);
    float*    pp2_32  = (float*)ws.take(NB * 128 * 4);
    _Float16* attnin  = (_Float16*)ws.take(NB * 384 * 2);      // [ctx | pre]
    float*    gi32    = (float*)ws.take(NB * 768 * 4);
    float*    gh32    = (float*)ws.take(NB * 768 * 4);
    float*    ah32    = (float*)ws.take(NB * 256 * 4);
    _Float16* ah16    = (_Float16*)ws.take(NB * 256 * 2);
    float*    q32     = (float*)ws.take(NB * 256 * 4);
    float*    u32     = (float*)ws.take(NB * TTEXT_ * 4);
    float*    sc32    = (float*)ws.take(NB * TTEXT_ * 4);
    _Float16* xxin    = (_Float16*)ws.take(NB * 512 * 2);      // [ctx2 | ah]
    float*    xx32    = (float*)ws.take(NB * 1024 * 4);
    _Float16* xx16    = (_Float16*)ws.take(NB * 1024 * 2);
    float*    gx4     = (float*)ws.take(NB * 4096 * 4);
    float*    gh4     = (float*)ws.take(NB * 4096 * 4);
    float*    h1_32   = (float*)ws.take(NB * 1024 * 4);
    float*    h2_32   = (float*)ws.take(NB * 1024 * 4);
    _Float16* h1_16   = (_Float16*)ws.take(NB * 1024 * 2);
    _Float16* h2_16   = (_Float16*)ws.take(NB * 1024 * 2);
    float*    c1_32   = (float*)ws.take(NB * 1024 * 4);
    float*    c2_32   = (float*)ws.take(NB * 1024 * 4);
    float*    xxa32   = (float*)ws.take(NB * 1024 * 4);
    _Float16* xxa16   = (_Float16*)ws.take(NB * 1024 * 2);
    float*    xxb32   = (float*)ws.take(NB * 1024 * 4);
    _Float16* xxb16   = (_Float16*)ws.take(NB * 1024 * 2);
    float*    mel32   = (float*)ws.take(NB * 1600 * 4);

    k_zero16<<<ceilDiv(NB * 384, 256), 256, 0, stream>>>(attnin, NB * 384);
    k_zero16<<<ceilDiv(NB * 512, 256), 256, 0, stream>>>(xxin, NB * 512);
    k_zero16<<<ceilDiv(NB * 256, 256), 256, 0, stream>>>(ah16, NB * 256);
    k_zero16<<<ceilDiv(NB * 1024, 256), 256, 0, stream>>>(h1_16, NB * 1024);
    k_zero16<<<ceilDiv(NB * 1024, 256), 256, 0, stream>>>(h2_16, NB * 1024);
    k_zero32<<<ceilDiv(NB * 256, 256), 256, 0, stream>>>(ah32, NB * 256);
    k_zero32<<<ceilDiv(NB * 1024, 256), 256, 0, stream>>>(c1_32, NB * 1024);
    k_zero32<<<ceilDiv(NB * 1024, 256), 256, 0, stream>>>(c2_32, NB * 1024);

    float* outMel  = (float*)d_out;
    float* outAttn = (float*)d_out + ATTN_OFF;

    // ---------------- decoder loop (400 steps) ----------------
    for (int step = 0; step < NSTEPS_; ++step) {
        int col = step * RVAL_ - 1;
        k_frame<<<ceilDiv(NB * NMELS_, 256), 256, 0, stream>>>(m, pin16, col < 0 ? 0 : col,
                                                               step == 0 ? 1 : 0);
        // decoder prenet
        gemm(stream, pin16, 80, dpW1, dp_b1, pp1_32, 256, pp1_16, 256, NB, 256, 80, ACT_RELU);
        gemm(stream, pp1_16, 256, dpW2, dp_b2, pp2_32, 128, attnin + 256, 384, NB, 128, 256, ACT_RELU);
        // attention GRU:  x = [ctx_prev | pre]
        gemm(stream, attnin, 384, arWih16, ar_bih, gi32, 768, nullptr, 0, NB, 768, 384, ACT_NONE);
        gemm(stream, ah16, 256, arWhh16, ar_bhh, gh32, 768, nullptr, 0, NB, 768, 256, ACT_NONE);
        k_gru_cell<<<ceilDiv(NB * 256, 256), 256, 0, stream>>>(gi32, gh32, ah32,
                                                               ah16, 256, xxin + 256, 512, 256);
        // attention
        gemm(stream, ah16, 256, attnW16, nullptr, q32, 256, nullptr, 0, NB, 256, 256, ACT_NONE);
        k_attn_score<<<dim3(TTEXT_, NB), 256, 0, stream>>>(ep32, q32, attn_v, u32, TTEXT_, 256);
        k_softmax<<<NB, 256, 0, stream>>>(u32, sc32, outAttn + (size_t)step * TTEXT_,
                                          TTEXT_, (size_t)NSTEPS_ * TTEXT_);
        k_context<<<NB, 256, 0, stream>>>(sc32, enc32, attnin, 384, xxin, 512, TTEXT_, 256);
        // rnn input projection + 2 residual LSTMs
        gemm(stream, xxin, 512, rnninW16, rnnin_b, xx32, 1024, xx16, 1024, NB, 1024, 512, ACT_NONE);
        gemm(stream, xx16, 1024, l1Wih16, l1_bih, gx4, 4096, nullptr, 0, NB, 4096, 1024, ACT_NONE);
        gemm(stream, h1_16, 1024, l1Whh16, l1_bhh, gh4, 4096, nullptr, 0, NB, 4096, 1024, ACT_NONE);
        k_lstm<<<ceilDiv(NB * 1024, 256), 256, 0, stream>>>(gx4, gh4, c1_32, h1_32, h1_16, 1024);
        k_add<<<ceilDiv(NB * 1024, 256), 256, 0, stream>>>(xx32, h1_32, xxa32, xxa16, NB * 1024);
        gemm(stream, xxa16, 1024, l2Wih16, l2_bih, gx4, 4096, nullptr, 0, NB, 4096, 1024, ACT_NONE);
        gemm(stream, h2_16, 1024, l2Whh16, l2_bhh, gh4, 4096, nullptr, 0, NB, 4096, 1024, ACT_NONE);
        k_lstm<<<ceilDiv(NB * 1024, 256), 256, 0, stream>>>(gx4, gh4, c2_32, h2_32, h2_16, 1024);
        k_add<<<ceilDiv(NB * 1024, 256), 256, 0, stream>>>(xxa32, h2_32, xxb32, xxb16, NB * 1024);
        // mel projection + scatter r frames
        gemm(stream, xxb16, 1024, melW16, nullptr, mel32, 1600, nullptr, 0, NB, 1600, 1024, ACT_NONE);
        k_write_mel<<<ceilDiv(NB * NMELS_ * RVAL_, 256), 256, 0, stream>>>(mel32, outMel, step);
    }

    // ---------------- postnet ----------------
    float*    pn32 = (float*)ws.take((size_t)NB * STEPS_ * 256 * 4);
    _Float16* pn16 = (_Float16*)ws.take((size_t)NB * STEPS_ * 256 * 2);
    run_cbhg(stream, ws, Pn, outMel, STEPS_, pn32, pn16);

    float* lin32 = (float*)ws.take((size_t)NB * STEPS_ * FFT_ * 4);
    gemm(stream, pn16, 256, postProj16, nullptr, lin32, FFT_, nullptr, 0,
         NB * STEPS_, FFT_, 256, ACT_NONE);
    k_lin_tr<<<ceilDiv(NB * STEPS_ * FFT_, 256), 256, 0, stream>>>(lin32, (float*)d_out + MEL_N);
}
